// NoVSkip_20547123544825
// MI455X (gfx1250) — matrix operand order repaired and checked by reading, not compiled, not run on hardware
//
#include <hip/hip_runtime.h>

// ---------------------------------------------------------------------------
// MI455X (gfx1250) implementation. wave32. WMMA bf16 -> f32 accumulate.
// All GEMM A/B operands pre-converted to bf16 once (no per-tile cvt VALU).
// Dims: V=20000, D=256, B=64, S=100, C=30, H=32
// ---------------------------------------------------------------------------

#define V_ 20000
#define D_ 256
#define B_ 64
#define S_ 100
#define C_ 30
#define H_ 32

typedef __attribute__((ext_vector_type(16))) __bf16 v16bf;
typedef __attribute__((ext_vector_type(8)))  float  v8f;

union FragU {
  v16bf v;
  unsigned short us[16];
  uint4 q[2];
};

__device__ inline unsigned short f2bf(float f) {
  unsigned u = __float_as_uint(f);
  return (unsigned short)((u + 0x7fffu + ((u >> 16) & 1u)) >> 16);
}

__device__ inline v8f zero8() {
  v8f z = {0.f, 0.f, 0.f, 0.f, 0.f, 0.f, 0.f, 0.f};
  return z;
}

__device__ inline v8f wmma_bf16(v16bf a, v16bf b, v8f c) {
  return __builtin_amdgcn_wmma_f32_16x16x32_bf16(false, a, false, b, (short)0, c,
                                                 false, false);
}

// A fragment (16x32 bf16), A stored row-major bf16.
// lanes 0-15 (kh=0): e<8 -> K=e,   e>=8 -> K=16+(e-8)
// lanes 16-31(kh=1): e<8 -> K=8+e, e>=8 -> K=24+(e-8)
// -> two contiguous 16B loads per lane.
__device__ inline v16bf a_frag_bf16(const unsigned short* row, int kb, int kh) {
  FragU f;
  const unsigned short* p0 = row + kb + 8 * kh;
  f.q[0] = *(const uint4*)(p0);
  f.q[1] = *(const uint4*)(p0 + 16);
  return f.v;
}

// B fragment (32x16 bf16) from packed-transposed weights (row n = output col n).
// lanes 0-15: e<8 -> K=e, e>=8 -> K=16+(e-8); lanes 16-31: e<8 -> K=8+e, e>=8 -> K=24+(e-8).
__device__ inline v16bf b_frag_bf16(const unsigned short* rowT, int kb, int kh) {
  FragU f;
  const unsigned short* p = rowT + kb + 8 * kh;
  f.q[0] = *(const uint4*)(p);
  f.q[1] = *(const uint4*)(p + 16);
  return f.v;
}

// One 16x16 tile: acc += A[m0:, 0:32*nch] * BT[n0:, kb0:kb0+32*nch]^T
__device__ inline v8f mm_tiles(const unsigned short* A, int lda,
                               const unsigned short* BT, int ldbt, int m0, int n0,
                               int kb0, int nch, v8f acc, int lane) {
  int r = lane & 15, kh = lane >> 4;
  const unsigned short* arow = A + (size_t)(m0 + r) * lda;
  const unsigned short* brow = BT + (size_t)(n0 + r) * ldbt;
  for (int c = 0; c < nch; ++c) {
    v16bf a = a_frag_bf16(arow, c * 32, kh);
    v16bf b = b_frag_bf16(brow, kb0 + c * 32, kh);
    acc = wmma_bf16(a, b, acc);
  }
  return acc;
}

__device__ inline void store_raw(float* C, int ldc, int m0, int n0, v8f acc, int lane) {
  int n = n0 + (lane & 15);
  int mb = m0 + ((lane >> 4) << 3);
#pragma unroll
  for (int rr = 0; rr < 8; ++rr) C[(size_t)(mb + rr) * ldc + n] = acc[rr];
}

__device__ inline float sigf(float x) { return 1.f / (1.f + expf(-x)); }

__device__ inline float wave_sum(float x) {
#pragma unroll
  for (int off = 16; off >= 1; off >>= 1) x += __shfl_xor(x, off, 32);
  return x;
}

// Row LayerNorm over 64 rows of 256, one wave per row (strided by 32 waves).
// Optionally writes a bf16 mirror for downstream WMMA consumption.
__device__ inline void ln_rows(const float* src, float* dst, unsigned short* dstb,
                               const float* g, const float* be, int wv, int lane) {
  for (int row = wv; row < 64; row += 32) {
    const float* s = src + row * 256;
    float v[8];
    float sum = 0.f;
#pragma unroll
    for (int e = 0; e < 8; ++e) {
      v[e] = s[lane * 8 + e];
      sum += v[e];
    }
    float mean = wave_sum(sum) * (1.f / 256.f);
    float var = 0.f;
#pragma unroll
    for (int e = 0; e < 8; ++e) {
      float d = v[e] - mean;
      var += d * d;
    }
    var = wave_sum(var) * (1.f / 256.f);
    float rstd = rsqrtf(var + 1e-5f);
#pragma unroll
    for (int e = 0; e < 8; ++e) {
      int n = lane * 8 + e;
      float o = (v[e] - mean) * rstd * g[n] + be[n];
      dst[row * 256 + n] = o;
      if (dstb) dstb[row * 256 + n] = f2bf(o);
    }
  }
}

__device__ inline void gsync() {
  __threadfence_block();
  __syncthreads();
}

// ---------------------------------------------------------------------------
// pack fp32 (KxN) weight -> bf16 transposed (NxK)
// ---------------------------------------------------------------------------
__global__ void packT_kernel(const float* __restrict__ W,
                             unsigned short* __restrict__ out, int K, int N) {
  int idx = blockIdx.x * blockDim.x + threadIdx.x;
  if (idx >= K * N) return;
  int k = idx / N, n = idx % N;
  out[(size_t)n * K + k] = f2bf(W[idx]);
}

// flat fp32 -> bf16
__global__ void cvt_kernel(const float* __restrict__ x, unsigned short* __restrict__ y,
                           int n) {
  int i = blockIdx.x * blockDim.x + threadIdx.x;
  if (i < n) y[i] = f2bf(x[i]);
}

// ---------------------------------------------------------------------------
// precompute constant vectors: cqc (tl-part of W_cq + b_cq), gc (tgt part of W_g1)
// ---------------------------------------------------------------------------
__global__ void pre_kernel(const float* tgt, const float* W_tl, const float* b_tl,
                           const float* W_cq, const float* b_cq, const float* W_g1,
                           const float* b_g1, float* cqc, float* gc) {
  __shared__ float s_tl[256];
  int n = threadIdx.x;
  float acc = b_tl[n];
  for (int k = 0; k < 256; ++k) acc += tgt[k] * W_tl[k * 256 + n];
  s_tl[n] = acc;
  __syncthreads();
  float a2 = b_cq[n];
  for (int k = 0; k < 256; ++k) a2 += s_tl[k] * W_cq[(288 + k) * 256 + n];
  cqc[n] = a2;
  float a3 = b_g1[n];
  for (int k = 0; k < 256; ++k) a3 += tgt[k] * W_g1[(320 + k) * 256 + n];
  gc[n] = a3;
}

// ---------------------------------------------------------------------------
// time features  tf = 1 - tanh((t*W + b)^2)  -> bf16 (GEMM-A only consumers)
// ---------------------------------------------------------------------------
__global__ void time_kernel(const float* st, const float* W_t1, const float* b_t1,
                            const float* W_t2, const float* b_t2, unsigned short* tf1b,
                            unsigned short* tf2b) {
  int idx = blockIdx.x * blockDim.x + threadIdx.x;
  if (idx >= B_ * S_ * 64) return;
  int row = idx >> 6, j = idx & 63;
  float t = st[row] * (1.f / 180.f);
  float u1 = t * W_t1[j] + b_t1[j];
  tf1b[idx] = f2bf(1.f - tanhf(u1 * u1));
  float u2 = t * W_t2[j] + b_t2[j];
  tf2b[idx] = f2bf(1.f - tanhf(u2 * u2));
}

// ---------------------------------------------------------------------------
// generic grid WMMA GEMM, split-K concat A (both bf16):
//   C[M,N] = act( A0[M,K0] ++ A1[M,K1]  @  BT^T  + cvec )
// lda1==0 broadcasts one A1 row. act: 0 none, 1 relu, 2 tanh.
// ---------------------------------------------------------------------------
__global__ void gemm2_kernel(float* C, int ldc, int M, int N,
                             const unsigned short* A0, int lda0, int K0,
                             const unsigned short* A1, int lda1, int K1,
                             const unsigned short* BT, int ldbt, const float* cvec,
                             int act, int rowClamp) {
  int lane = threadIdx.x & 31;
  int wid = blockIdx.x * (blockDim.x >> 5) + (threadIdx.x >> 5);
  int nt = N >> 4;
  int total = (M >> 4) * nt;
  if (wid >= total) return;
  int m0 = (wid / nt) << 4, n0 = (wid % nt) << 4;
  int r = lane & 15, kh = lane >> 4;
  int row = m0 + r;
  if (rowClamp >= 0 && row > rowClamp) row = rowClamp;
  const unsigned short* brow = BT + (size_t)(n0 + r) * ldbt;
  v8f acc = zero8();
  const unsigned short* a0row = A0 + (size_t)row * lda0;
  for (int c = 0; c < (K0 >> 5); ++c)
    acc = wmma_bf16(a_frag_bf16(a0row, c * 32, kh), b_frag_bf16(brow, c * 32, kh), acc);
  if (K1 > 0) {
    const unsigned short* a1row = A1 + (size_t)row * lda1;
    for (int c = 0; c < (K1 >> 5); ++c)
      acc = wmma_bf16(a_frag_bf16(a1row, c * 32, kh),
                      b_frag_bf16(brow, K0 + c * 32, kh), acc);
  }
  int n = n0 + (lane & 15);
  int mb = m0 + (kh << 3);
  float cv = cvec ? cvec[n] : 0.f;
#pragma unroll
  for (int rr = 0; rr < 8; ++rr) {
    float v = acc[rr] + cv;
    if (act == 1) v = fmaxf(v, 0.f);
    else if (act == 2) v = tanhf(v);
    C[(size_t)(mb + rr) * ldc + n] = v;
  }
}

// ---------------------------------------------------------------------------
// gumbel hard code-selection mask from H1 = tanh(cat@W_cs1+b)
// ---------------------------------------------------------------------------
__global__ void mask_kernel(const float* H1, const float* W_cs2, const float* b_cs2,
                            const float* gum, float* maskv) {
  int v = blockIdx.x * blockDim.x + threadIdx.x;
  if (v > V_) return;
  const float* h = H1 + (size_t)v * 256;
  float l0 = b_cs2[0], l1 = b_cs2[1];
  for (int n = 0; n < 256; ++n) {
    float x = h[n];
    l0 += x * W_cs2[2 * n];
    l1 += x * W_cs2[2 * n + 1];
  }
  maskv[v] = (l0 + gum[2 * v] >= l1 + gum[2 * v + 1]) ? 1.f : 0.f;
}

// ---------------------------------------------------------------------------
// selected[b,s,:] = sum_c mask(v)*relu(emb[v]) + time_enc -> bf16 (LSTM A only)
// ---------------------------------------------------------------------------
__global__ void select_kernel(const int* seq, const float* emb, const float* maskv,
                              const float* te, unsigned short* selb) {
  int row = blockIdx.x;  // b*S + s
  int d = threadIdx.x;   // 0..255
  float acc = te[(size_t)row * 256 + d];
  const int* sp = seq + (size_t)row * C_;
  for (int c = 0; c < C_; ++c) {
    int v = sp[c];
    if (maskv[v] != 0.f) acc += fmaxf(emb[(size_t)v * 256 + d], 0.f);
  }
  selb[(size_t)row * 256 + d] = f2bf(acc);
}

// ---------------------------------------------------------------------------
// reversed LSTM, persistent single workgroup (16 waves), WMMA gates.
// ---------------------------------------------------------------------------
__global__ void __launch_bounds__(512) lstm_kernel(const unsigned short* selb,
                                                   const unsigned short* WTih,
                                                   const unsigned short* WThh,
                                                   unsigned short* ffb) {
  __shared__ __align__(16) float s_g[64 * 128];
  __shared__ float s_c[64 * 32];
  __shared__ __align__(16) unsigned short s_hb[64 * 32];
  int tid = threadIdx.x, lane = tid & 31, wv = tid >> 5;
  for (int j = tid; j < 64 * 32; j += 512) {
    s_c[j] = 0.f;
    s_hb[j] = 0;
  }
  __syncthreads();
  for (int t = 0; t < S_; ++t) {
    int s = S_ - 1 - t;  // reversed scan
    for (int tile = wv; tile < 32; tile += 16) {
      int m0 = (tile >> 3) << 4, n0 = (tile & 7) << 4;
      int r = lane & 15, kh = lane >> 4;
      v8f acc = zero8();
      const unsigned short* arow = selb + ((size_t)(m0 + r) * S_ + s) * 256;
      const unsigned short* brow = WTih + (size_t)(n0 + r) * 256;
      for (int c = 0; c < 8; ++c)
        acc = wmma_bf16(a_frag_bf16(arow, c * 32, kh), b_frag_bf16(brow, c * 32, kh),
                        acc);
      const unsigned short* hrow = s_hb + (m0 + r) * 32;
      const unsigned short* brow2 = WThh + (size_t)(n0 + r) * 32;
      acc = wmma_bf16(a_frag_bf16(hrow, 0, kh), b_frag_bf16(brow2, 0, kh), acc);
      store_raw(s_g, 128, m0, n0, acc, lane);
    }
    __syncthreads();
    for (int j = tid; j < 64 * 32; j += 512) {
      int b = j >> 5, hh = j & 31;
      float gi = s_g[b * 128 + hh];
      float gf = s_g[b * 128 + 32 + hh];
      float gg = s_g[b * 128 + 64 + hh];
      float go = s_g[b * 128 + 96 + hh];
      float c = sigf(gf) * s_c[j] + sigf(gi) * tanhf(gg);
      float h = sigf(go) * tanhf(c);
      s_c[j] = c;
      unsigned short hb = f2bf(h);
      s_hb[j] = hb;
      ffb[(size_t)s * 64 * 32 + j] = hb;
    }
    __syncthreads();
  }
}

// ---------------------------------------------------------------------------
// Persistent scan kernel: 100-step recurrent core. 1 workgroup, 32 waves.
// mem rows computed incrementally (O(S) not O(S^2) GEMMs).
// ---------------------------------------------------------------------------
struct ScanArgs {
  const int* seq;
  const float* emb;
  const float* maskv;
  const float* te;          // [B*S,256] fp32
  const unsigned short* ffb;  // [S,64,32] bf16
  const float* cqc;           // [256]
  const unsigned short *wt_cq, *wt_fc, *wt_ihg, *wt_hhg, *wt_step, *wt_mem, *wt_f1,
      *wt_f2;
  const float *b_fc, *b_step, *g_step, *be_step, *g_ln2, *b_ln2, *g_ln, *b_ln, *b_f1,
      *b_f2, *b_mem, *g_mem, *be_mem;
  // fp32 state
  float *hx, *cq, *shx, *sh, *tmp, *hx2, *gi, *gh, *mem, *hid;
  // bf16 mirrors (GEMM A operands)
  unsigned short *hxb, *z0b, *zb, *shxb, *hx2b, *f1bb, *hidb;
};

__global__ void __launch_bounds__(1024) scan_kernel(ScanArgs A) {
  const int tid = threadIdx.x, lane = tid & 31, wv = tid >> 5;  // 32 waves
  __shared__ float s_e[64 * 32];
  __shared__ float s_pm[64 * 32];
  __shared__ int s_vi[64 * 32];
  __shared__ float s_me[64 * S_];

  for (int j = tid; j < 64 * 256; j += 1024) {
    A.hx[j] = 0.f;
    A.hxb[j] = 0;  // bf16 +0.0
  }
  gsync();

  for (int i = 0; i < S_; ++i) {
    // ---- P1: cq = relu(hx@Wcq[0:256] + ff_i@Wcq[256:288] + cqc) ----
    for (int t = wv; t < 64; t += 32) {
      int m0 = (t >> 4) << 4, n0 = (t & 15) << 4;
      v8f acc = zero8();
      acc = mm_tiles(A.hxb, 256, A.wt_cq, 544, m0, n0, 0, 8, acc, lane);
      acc = mm_tiles(A.ffb + (size_t)i * 64 * 32, 32, A.wt_cq, 544, m0, n0, 256, 1, acc,
                     lane);
      int n = n0 + (lane & 15), mb = m0 + ((lane >> 4) << 3);
      float cv = A.cqc[n];
#pragma unroll
      for (int rr = 0; rr < 8; ++rr)
        A.cq[(size_t)(mb + rr) * 256 + n] = fmaxf(acc[rr] + cv, 0.f);
    }
    gsync();
    // ---- P2: attention over codes ----
    for (int p = tid; p < 64 * C_; p += 1024) {
      int b = p / C_, c = p % C_;
      int v = A.seq[((size_t)b * S_ + i) * C_ + c];
      const float4* er = (const float4*)(A.emb + (size_t)v * 256);
      const float4* cqr = (const float4*)(A.cq + b * 256);
      float acc = 0.f;
      for (int d = 0; d < 64; ++d) {
        float4 cv = cqr[d];
        float4 ev = er[d];
        acc += cv.x * fmaxf(ev.x, 0.f) + cv.y * fmaxf(ev.y, 0.f) +
               cv.z * fmaxf(ev.z, 0.f) + cv.w * fmaxf(ev.w, 0.f);
      }
      s_e[b * 32 + c] = acc * (1.f / 16.f);
      s_pm[b * 32 + c] = A.maskv[v];
      s_vi[b * 32 + c] = v;
    }
    __syncthreads();
    if (tid < 64) {
      int b = tid;
      float mx = -1e30f;
      for (int c = 0; c < C_; ++c) mx = fmaxf(mx, s_e[b * 32 + c]);
      float sum = 0.f;
      for (int c = 0; c < C_; ++c) {
        float ex = expf(s_e[b * 32 + c] - mx);
        s_e[b * 32 + c] = ex;
        sum += ex;
      }
      float inv = 1.f / sum;
      for (int c = 0; c < C_; ++c) s_e[b * 32 + c] *= inv * s_pm[b * 32 + c];
    }
    __syncthreads();
    // ---- P2c: z0 = att @ relu(emb)  (bf16 mirror only; GEMM-only consumer) ----
    for (int p = tid; p < 64 * 64; p += 1024) {  // (b, d4) pairs, 4 floats each
      int b = p >> 6, d4 = (p & 63);
      float a0 = 0.f, a1 = 0.f, a2 = 0.f, a3 = 0.f;
      for (int c = 0; c < C_; ++c) {
        float w = s_e[b * 32 + c];
        float4 ev = *(const float4*)(A.emb + (size_t)s_vi[b * 32 + c] * 256 + d4 * 4);
        a0 += w * fmaxf(ev.x, 0.f);
        a1 += w * fmaxf(ev.y, 0.f);
        a2 += w * fmaxf(ev.z, 0.f);
        a3 += w * fmaxf(ev.w, 0.f);
      }
      unsigned short* o = A.z0b + (size_t)b * 256 + d4 * 4;
      o[0] = f2bf(a0); o[1] = f2bf(a1); o[2] = f2bf(a2); o[3] = f2bf(a3);
    }
    gsync();
    // ---- P3: z = z0@W_fc + b_fc + te_i  (bf16 only; feeds GRU GEMM) ----
    for (int t = wv; t < 64; t += 32) {
      int m0 = (t >> 4) << 4, n0 = (t & 15) << 4;
      v8f acc = mm_tiles(A.z0b, 256, A.wt_fc, 256, m0, n0, 0, 8, zero8(), lane);
      int n = n0 + (lane & 15), mb = m0 + ((lane >> 4) << 3);
#pragma unroll
      for (int rr = 0; rr < 8; ++rr) {
        int b = mb + rr;
        A.zb[(size_t)b * 256 + n] =
            f2bf(acc[rr] + A.b_fc[n] + A.te[((size_t)b * S_ + i) * 256 + n]);
      }
    }
    gsync();
    // ---- P4: GRU gate GEMMs (z@W_ih_g, hx@W_hh_g) ----
    for (int t = wv; t < 384; t += 32) {
      int which = (t >= 192);
      int tt = which ? t - 192 : t;
      int m0 = (tt / 48) << 4, n0 = (tt % 48) << 4;
      const unsigned short* Asrc = which ? A.hxb : A.zb;
      const unsigned short* W = which ? A.wt_hhg : A.wt_ihg;
      float* Cd = which ? A.gh : A.gi;
      v8f acc = mm_tiles(Asrc, 256, W, 256, m0, n0, 0, 8, zero8(), lane);
      store_raw(Cd, 768, m0, n0, acc, lane);
    }
    gsync();
    // ---- P5: GRU elementwise ----
    for (int j = tid; j < 64 * 256; j += 1024) {
      int b = j >> 8, d = j & 255;
      float r_ = sigf(A.gi[(size_t)b * 768 + d] + A.gh[(size_t)b * 768 + d]);
      float zz = sigf(A.gi[(size_t)b * 768 + 256 + d] + A.gh[(size_t)b * 768 + 256 + d]);
      float n_ =
          tanhf(A.gi[(size_t)b * 768 + 512 + d] + r_ * A.gh[(size_t)b * 768 + 512 + d]);
      float sx = (1.f - zz) * n_ + zz * A.hx[j];
      A.shx[j] = sx;
      A.shxb[j] = f2bf(sx);
    }
    gsync();
    // ---- P6: sh = LN(relu(step_hx@W_step + b_step)) ----
    for (int t = wv; t < 64; t += 32) {
      int m0 = (t >> 4) << 4, n0 = (t & 15) << 4;
      v8f acc = mm_tiles(A.shxb, 256, A.wt_step, 256, m0, n0, 0, 8, zero8(), lane);
      int n = n0 + (lane & 15), mb = m0 + ((lane >> 4) << 3);
#pragma unroll
      for (int rr = 0; rr < 8; ++rr)
        A.tmp[(size_t)(mb + rr) * 256 + n] = fmaxf(acc[rr] + A.b_step[n], 0.f);
    }
    gsync();
    ln_rows(A.tmp, A.sh, nullptr, A.g_step, A.be_step, wv, lane);
    gsync();

    if (i > 0) {
      // ---- P7: memory attention over past hiddens (j < i; rest masked out) ----
      int tot = i * 64;
      for (int p = tid; p < tot; p += 1024) {
        int jrow = p >> 6, b = p & 63;
        const float4* s = (const float4*)(A.sh + b * 256);
        const float4* m = (const float4*)(A.mem + ((size_t)jrow * 64 + b) * 256);
        float acc = 0.f;
        for (int d = 0; d < 64; ++d) {
          float4 sv = s[d];
          float4 mv = m[d];
          acc += sv.x * mv.x + sv.y * mv.y + sv.z * mv.z + sv.w * mv.w;
        }
        s_me[b * S_ + jrow] = acc * (1.f / 16.f);
      }
      __syncthreads();
      if (tid < 64) {
        int b = tid;
        float mx = -1e30f;
        for (int j2 = 0; j2 < i; ++j2) mx = fmaxf(mx, s_me[b * S_ + j2]);
        float sum = 0.f;
        for (int j2 = 0; j2 < i; ++j2) {
          float ex = expf(s_me[b * S_ + j2] - mx);
          s_me[b * S_ + j2] = ex;
          sum += ex;
        }
        float inv = 1.f / sum;
        for (int j2 = 0; j2 < i; ++j2) s_me[b * S_ + j2] *= inv;
      }
      __syncthreads();
      for (int p = tid; p < 64 * 64; p += 1024) {  // (b, d4)
        int b = p >> 6, d4 = (p & 63);
        float a0 = 0.f, a1 = 0.f, a2 = 0.f, a3 = 0.f;
        for (int j2 = 0; j2 < i; ++j2) {
          float w = s_me[b * S_ + j2];
          float4 mv = *(const float4*)(A.mem + ((size_t)j2 * 64 + b) * 256 + d4 * 4);
          a0 += w * mv.x; a1 += w * mv.y; a2 += w * mv.z; a3 += w * mv.w;
        }
        float* o = A.tmp + (size_t)b * 256 + d4 * 4;
        const float* sr = A.sh + (size_t)b * 256 + d4 * 4;
        o[0] = a0 + sr[0]; o[1] = a1 + sr[1]; o[2] = a2 + sr[2]; o[3] = a3 + sr[3];
      }
      gsync();
      ln_rows(A.tmp, A.hx2, A.hx2b, A.g_ln2, A.b_ln2, wv, lane);
      gsync();
      // ---- P8: FFN + residual + LN ----
      for (int t = wv; t < 256; t += 32) {  // 4 x 64 tiles (N=1024)
        int m0 = (t / 64) << 4, n0 = (t % 64) << 4;
        v8f acc = mm_tiles(A.hx2b, 256, A.wt_f1, 256, m0, n0, 0, 8, zero8(), lane);
        int n = n0 + (lane & 15), mb = m0 + ((lane >> 4) << 3);
#pragma unroll
        for (int rr = 0; rr < 8; ++rr)
          A.f1bb[(size_t)(mb + rr) * 1024 + n] = f2bf(fmaxf(acc[rr] + A.b_f1[n], 0.f));
      }
      gsync();
      for (int t = wv; t < 64; t += 32) {
        int m0 = (t >> 4) << 4, n0 = (t & 15) << 4;
        v8f acc = mm_tiles(A.f1bb, 1024, A.wt_f2, 1024, m0, n0, 0, 32, zero8(), lane);
        int n = n0 + (lane & 15), mb = m0 + ((lane >> 4) << 3);
#pragma unroll
        for (int rr = 0; rr < 8; ++rr)
          A.tmp[(size_t)(mb + rr) * 256 + n] =
              acc[rr] + A.b_f2[n] + A.hx2[(size_t)(mb + rr) * 256 + n];
      }
      gsync();
      ln_rows(A.tmp, A.hx2, A.hx2b, A.g_ln, A.b_ln, wv, lane);
      gsync();
    }
    // ---- P9: hx_new select + hidden store (fp32 + bf16 mirrors) ----
    for (int j = tid; j < 64 * 256; j += 1024) {
      float h = (i == 0) ? A.shx[j] : A.hx2[j];
      A.hx[j] = h;
      unsigned short hb = f2bf(h);
      A.hxb[j] = hb;
      int b = j >> 8, d = j & 255;
      A.hid[((size_t)b * S_ + i) * 256 + d] = h;
      A.hidb[((size_t)b * S_ + i) * 256 + d] = hb;
    }
    gsync();
    // ---- P10: mem[i] = LN(relu(hx_new@W_mem + b_mem)) (incremental!) ----
    for (int t = wv; t < 64; t += 32) {
      int m0 = (t >> 4) << 4, n0 = (t & 15) << 4;
      v8f acc = mm_tiles(A.hxb, 256, A.wt_mem, 256, m0, n0, 0, 8, zero8(), lane);
      int n = n0 + (lane & 15), mb = m0 + ((lane >> 4) << 3);
#pragma unroll
      for (int rr = 0; rr < 8; ++rr)
        A.tmp[(size_t)(mb + rr) * 256 + n] = fmaxf(acc[rr] + A.b_mem[n], 0.f);
    }
    gsync();
    ln_rows(A.tmp, A.mem + (size_t)i * 64 * 256, nullptr, A.g_mem, A.be_mem, wv, lane);
    gsync();
  }
}

// ---------------------------------------------------------------------------
// Output stage kernels
// ---------------------------------------------------------------------------
__global__ void ge_kernel(const float* h1g, const float* W_g2, const float* b_g2,
                          const int* lengths, float* ge) {
  int row = blockIdx.x * blockDim.x + threadIdx.x;
  if (row >= B_ * S_) return;
  int b = row / S_, s = row % S_;
  float acc = b_g2[0];
  const float* h = h1g + (size_t)row * 256;
  for (int n = 0; n < 256; ++n) acc += h[n] * W_g2[n];
  ge[row] = (s >= lengths[b]) ? -1e9f : acc;
}

__global__ void out_kernel(const float* ge, const float* hid, const float* W_o,
                           const float* b_o, float* out) {
  int b = blockIdx.x;
  int d = threadIdx.x;
  __shared__ float s_w[S_];
  __shared__ float s_acc[256];
  if (d < S_) s_w[d] = ge[b * S_ + d];
  __syncthreads();
  if (d == 0) {
    float mx = -1e30f;
    for (int s = 0; s < S_; ++s) mx = fmaxf(mx, s_w[s]);
    float sum = 0.f;
    for (int s = 0; s < S_; ++s) {
      float e = expf(s_w[s] - mx);
      s_w[s] = e;
      sum += e;
    }
    float inv = 1.f / sum;
    for (int s = 0; s < S_; ++s) s_w[s] *= inv;
  }
  __syncthreads();
  float acc = 0.f;
  for (int s = 0; s < S_; ++s) acc += s_w[s] * hid[((size_t)b * S_ + s) * 256 + d];
  s_acc[d] = acc;
  __syncthreads();
  if (d < 2) {
    float o = b_o[d];
    for (int n = 0; n < 256; ++n) o += s_acc[n] * W_o[n * 2 + d];
    out[b * 2 + d] = o;
  }
}

// ---------------------------------------------------------------------------
// Host launcher
// ---------------------------------------------------------------------------
extern "C" void kernel_launch(void* const* d_in, const int* in_sizes, int n_in,
                              void* d_out, int out_size, void* d_ws, size_t ws_size,
                              hipStream_t stream) {
  (void)in_sizes; (void)n_in; (void)out_size; (void)ws_size;
  const int* seq = (const int*)d_in[0];
  const int* lengths = (const int*)d_in[1];
  const float* st = (const float*)d_in[2];
  const float* emb = (const float*)d_in[3];
  const float* tgt = (const float*)d_in[4];
  const float* W_tl = (const float*)d_in[5];
  const float* b_tl = (const float*)d_in[6];
  const float* W_cs1 = (const float*)d_in[7];
  const float* b_cs1 = (const float*)d_in[8];
  const float* W_cs2 = (const float*)d_in[9];
  const float* b_cs2 = (const float*)d_in[10];
  const float* W_ih_l = (const float*)d_in[11];
  const float* W_hh_l = (const float*)d_in[12];
  const float* W_cq = (const float*)d_in[13];
  const float* b_cq = (const float*)d_in[14];
  const float* W_fc = (const float*)d_in[15];
  const float* b_fc = (const float*)d_in[16];
  const float* W_ih_g = (const float*)d_in[17];
  const float* W_hh_g = (const float*)d_in[18];
  const float* W_mem = (const float*)d_in[19];
  const float* b_mem = (const float*)d_in[20];
  const float* g_mem = (const float*)d_in[21];
  const float* be_mem = (const float*)d_in[22];
  const float* W_step = (const float*)d_in[23];
  const float* b_step = (const float*)d_in[24];
  const float* g_step = (const float*)d_in[25];
  const float* be_step = (const float*)d_in[26];
  const float* g_ln2 = (const float*)d_in[27];
  const float* b_ln2 = (const float*)d_in[28];
  const float* g_ln = (const float*)d_in[29];
  const float* b_ln = (const float*)d_in[30];
  const float* W_f1 = (const float*)d_in[31];
  const float* b_f1 = (const float*)d_in[32];
  const float* W_f2 = (const float*)d_in[33];
  const float* b_f2 = (const float*)d_in[34];
  const float* W_t1 = (const float*)d_in[35];
  const float* b_t1 = (const float*)d_in[36];
  const float* W_up = (const float*)d_in[37];
  const float* b_up = (const float*)d_in[38];
  const float* W_t2 = (const float*)d_in[39];
  const float* b_t2 = (const float*)d_in[40];
  const float* W_g1 = (const float*)d_in[41];
  const float* b_g1 = (const float*)d_in[42];
  const float* W_g2 = (const float*)d_in[43];
  const float* b_g2 = (const float*)d_in[44];
  const float* W_o = (const float*)d_in[45];
  const float* b_o = (const float*)d_in[46];
  const float* gum = (const float*)d_in[47];

  char* ws = (char*)d_ws;
  size_t off = 0;
  auto alloc = [&](size_t bytes) -> void* {
    off = (off + 255) & ~(size_t)255;
    void* p = ws + off;
    off += bytes;
    return p;
  };
  // packed bf16 transposed weights
  unsigned short* wt_cs1 = (unsigned short*)alloc(256 * 512 * 2);
  unsigned short* wt_up = (unsigned short*)alloc(256 * 64 * 2);
  unsigned short* wt_ihl = (unsigned short*)alloc(128 * 256 * 2);
  unsigned short* wt_hhl = (unsigned short*)alloc(128 * 32 * 2);
  unsigned short* wt_cq = (unsigned short*)alloc(256 * 544 * 2);
  unsigned short* wt_fc = (unsigned short*)alloc(256 * 256 * 2);
  unsigned short* wt_ihg = (unsigned short*)alloc(768 * 256 * 2);
  unsigned short* wt_hhg = (unsigned short*)alloc(768 * 256 * 2);
  unsigned short* wt_mem = (unsigned short*)alloc(256 * 256 * 2);
  unsigned short* wt_step = (unsigned short*)alloc(256 * 256 * 2);
  unsigned short* wt_f1 = (unsigned short*)alloc(1024 * 256 * 2);
  unsigned short* wt_f2 = (unsigned short*)alloc(256 * 1024 * 2);
  unsigned short* wt_g1 = (unsigned short*)alloc(256 * 576 * 2);
  // bf16 activations / A-operands
  unsigned short* emb_bf = (unsigned short*)alloc((size_t)(V_ + 1) * 256 * 2);
  unsigned short* tgt_bf = (unsigned short*)alloc(256 * 2);
  unsigned short* tf1b = (unsigned short*)alloc((size_t)B_ * S_ * 64 * 2);
  unsigned short* tf2b = (unsigned short*)alloc((size_t)B_ * S_ * 64 * 2);
  unsigned short* selb = (unsigned short*)alloc((size_t)B_ * S_ * 256 * 2);
  unsigned short* ffbb = (unsigned short*)alloc((size_t)S_ * 64 * 32 * 2);
  unsigned short* hxb = (unsigned short*)alloc(64 * 256 * 2);
  unsigned short* z0b = (unsigned short*)alloc(64 * 256 * 2);
  unsigned short* zbb = (unsigned short*)alloc(64 * 256 * 2);
  unsigned short* shxb = (unsigned short*)alloc(64 * 256 * 2);
  unsigned short* hx2b = (unsigned short*)alloc(64 * 256 * 2);
  unsigned short* f1bb = (unsigned short*)alloc(64 * 1024 * 2);
  unsigned short* hidb = (unsigned short*)alloc((size_t)B_ * S_ * 256 * 2);
  // fp32 buffers
  float* te = (float*)alloc((size_t)B_ * S_ * 256 * 4);
  float* H1 = (float*)alloc((size_t)20016 * 256 * 4);
  float* maskv = (float*)alloc((size_t)20016 * 4);
  float* cqc = (float*)alloc(256 * 4);
  float* gc = (float*)alloc(256 * 4);
  float* hx = (float*)alloc(64 * 256 * 4);
  float* cq = (float*)alloc(64 * 256 * 4);
  float* shx = (float*)alloc(64 * 256 * 4);
  float* sh = (float*)alloc(64 * 256 * 4);
  float* tmp = (float*)alloc(64 * 256 * 4);
  float* hx2 = (float*)alloc(64 * 256 * 4);
  float* gi = (float*)alloc(64 * 768 * 4);
  float* gh = (float*)alloc(64 * 768 * 4);
  float* memb = (float*)alloc((size_t)S_ * 64 * 256 * 4);
  float* hid = (float*)alloc((size_t)B_ * S_ * 256 * 4);
  float* h1g = (float*)alloc((size_t)B_ * S_ * 256 * 4);
  float* geb = (float*)alloc((size_t)B_ * S_ * 4);

  auto pack = [&](const float* W, unsigned short* o, int K, int N) {
    int n = K * N;
    packT_kernel<<<(n + 255) / 256, 256, 0, stream>>>(W, o, K, N);
  };
  pack(W_cs1, wt_cs1, 512, 256);
  pack(W_up, wt_up, 64, 256);
  pack(W_ih_l, wt_ihl, 256, 128);
  pack(W_hh_l, wt_hhl, 32, 128);
  pack(W_cq, wt_cq, 544, 256);
  pack(W_fc, wt_fc, 256, 256);
  pack(W_ih_g, wt_ihg, 256, 768);
  pack(W_hh_g, wt_hhg, 256, 768);
  pack(W_mem, wt_mem, 256, 256);
  pack(W_step, wt_step, 256, 256);
  pack(W_f1, wt_f1, 256, 1024);
  pack(W_f2, wt_f2, 1024, 256);
  pack(W_g1, wt_g1, 576, 256);
  cvt_kernel<<<((V_ + 1) * 256 + 255) / 256, 256, 0, stream>>>(emb, emb_bf,
                                                               (V_ + 1) * 256);
  cvt_kernel<<<1, 256, 0, stream>>>(tgt, tgt_bf, 256);

  pre_kernel<<<1, 256, 0, stream>>>(tgt, W_tl, b_tl, W_cq, b_cq, W_g1, b_g1, cqc, gc);
  time_kernel<<<(B_ * S_ * 64 + 255) / 256, 256, 0, stream>>>(st, W_t1, b_t1, W_t2, b_t2,
                                                              tf1b, tf2b);
  // time_enc = tf1 @ W_up + b_up    (M=6400, K=64, N=256)
  {
    int waves = (B_ * S_ / 16) * (256 / 16);
    gemm2_kernel<<<(waves + 7) / 8, 256, 0, stream>>>(te, 256, B_ * S_, 256, tf1b, 64, 64,
                                                      nullptr, 0, 0, wt_up, 64, b_up, 0,
                                                      -1);
  }
  // H1 = tanh([emb | tgt] @ W_cs1 + b_cs1)  (M=20016 padded, K=256+256, N=256)
  {
    int waves = (20016 / 16) * (256 / 16);
    gemm2_kernel<<<(waves + 7) / 8, 256, 0, stream>>>(H1, 256, 20016, 256, emb_bf, 256,
                                                      256, tgt_bf, 0, 256, wt_cs1, 512,
                                                      b_cs1, 2, V_);
  }
  mask_kernel<<<(V_ + 1 + 255) / 256, 256, 0, stream>>>(H1, W_cs2, b_cs2, gum, maskv);
  select_kernel<<<B_ * S_, 256, 0, stream>>>(seq, emb, maskv, te, selb);
  lstm_kernel<<<1, 512, 0, stream>>>(selb, wt_ihl, wt_hhl, ffbb);

  ScanArgs sa;
  sa.seq = seq; sa.emb = emb; sa.maskv = maskv; sa.te = te; sa.ffb = ffbb; sa.cqc = cqc;
  sa.wt_cq = wt_cq; sa.wt_fc = wt_fc; sa.wt_ihg = wt_ihg; sa.wt_hhg = wt_hhg;
  sa.wt_step = wt_step; sa.wt_mem = wt_mem; sa.wt_f1 = wt_f1; sa.wt_f2 = wt_f2;
  sa.b_fc = b_fc; sa.b_step = b_step; sa.g_step = g_step; sa.be_step = be_step;
  sa.g_ln2 = g_ln2; sa.b_ln2 = b_ln2; sa.g_ln = g_ln; sa.b_ln = b_ln;
  sa.b_f1 = b_f1; sa.b_f2 = b_f2; sa.b_mem = b_mem; sa.g_mem = g_mem; sa.be_mem = be_mem;
  sa.hx = hx; sa.cq = cq; sa.shx = shx; sa.sh = sh; sa.tmp = tmp; sa.hx2 = hx2;
  sa.gi = gi; sa.gh = gh; sa.mem = memb; sa.hid = hid;
  sa.hxb = hxb; sa.z0b = z0b; sa.zb = zbb; sa.shxb = shxb; sa.hx2b = hx2b;
  sa.f1bb = f1bb; sa.hidb = hidb;
  scan_kernel<<<1, 1024, 0, stream>>>(sa);

  // h1g = tanh([hiddens | tf2] @ W_g1[0:320] + gc)
  {
    int waves = (B_ * S_ / 16) * (256 / 16);
    gemm2_kernel<<<(waves + 7) / 8, 256, 0, stream>>>(h1g, 256, B_ * S_, 256, hidb, 256,
                                                      256, tf2b, 64, 64, wt_g1, 576, gc,
                                                      2, -1);
  }
  ge_kernel<<<(B_ * S_ + 255) / 256, 256, 0, stream>>>(h1g, W_g2, b_g2, lengths, geb);
  out_kernel<<<B_, 256, 0, stream>>>(geb, hid, W_o, b_o, (float*)d_out);
}
